// GraphAttentionEncoder_28922309771787
// MI455X (gfx1250) — matrix operand order repaired and checked by reading, not compiled, not run on hardware
//
#include <hip/hip_runtime.h>
#include <hip/hip_bf16.h>
#include <cstdint>
#include <cmath>

// ---------------------------------------------------------------------------
// GAT 2-layer encoder for MI455X (gfx1250, wave32).
//  - GEMMs via v_wmma_f32_16x16x32_bf16. A/B are pre-packed once into bf16
//    WMMA-fragment order, so the GEMM inner loop is 2x 32B vector loads +
//    1 WMMA per k-step (no cvt, no strided gathers in the hot loop).
//  - Edge softmax/scatter via native f32 atomics (L2-resident: h / agg
//    buffers are 51.2 MB each, well inside the 192 MB L2).
// ---------------------------------------------------------------------------

#define NN   50000
#define EE   800000
#define HEADS1 4
#define CCH  64
#define SLOPE 0.2f

typedef __attribute__((ext_vector_type(16))) __bf16 v16bf;
typedef __attribute__((ext_vector_type(8)))  float  v8f;

// ---------------- fill ----------------
__global__ void fill_f32(float* __restrict__ p, float v, int n) {
  for (int i = blockIdx.x * blockDim.x + threadIdx.x; i < n;
       i += gridDim.x * blockDim.x)
    p[i] = v;
}

// ---------------- pack A[M,K] f32 -> bf16 in A-fragment order ---------------
// per row: dest offset kt*32 + h*16 + j  holds  src K = kt*32 + h*16 + j
__global__ void pack_A_bf16(const float* __restrict__ in,
                            __bf16* __restrict__ out, int M, int K) {
  int idx = blockIdx.x * blockDim.x + threadIdx.x;
  if (idx >= M * K) return;
  int r = idx / K, o = idx - r * K;
  int kt = o >> 5, rem = o & 31, h = rem >> 4, j = rem & 15;
  int ks = (kt << 5) + h * 16 + j;
  out[idx] = (__bf16)in[(size_t)r * K + ks];
}

// ---------------- pack B[K,Nc] f32 -> bf16 column-major (B-fragment order) --
// out[n*K + k] = in[k*Nc + n]  (this code's k order: k = 16h + s, same as packed A)
__global__ void pack_B_bf16(const float* __restrict__ in,
                            __bf16* __restrict__ out, int K, int Nc) {
  int idx = blockIdx.x * blockDim.x + threadIdx.x;
  if (idx >= K * Nc) return;
  int col = idx / K, k = idx - col * K;
  out[idx] = (__bf16)in[(size_t)k * Nc + col];
}

// ---------------- WMMA GEMM on packed operands ------------------------------
// C[M,Nc](f32) = A[M,K] * B[K,Nc]; one wave = one 16x16 C tile.
__global__ __launch_bounds__(256) void gemm_bf16_wmma(
    const __bf16* __restrict__ Apack, const __bf16* __restrict__ Bpack,
    float* __restrict__ C, int M, int K, int Nc) {
  const int lane = threadIdx.x & 31;
  const int hlf  = lane >> 4;   // lane group 0/1
  const int lrow = lane & 15;
  const int wave = blockIdx.x * (blockDim.x >> 5) + (threadIdx.x >> 5);
  const int tilesN = Nc >> 4;
  const int tm = (wave / tilesN) << 4;
  const int tn = (wave % tilesN) << 4;
  if (tm >= M) return;

  const __bf16* Ap = Apack + (size_t)(tm + lrow) * K + hlf * 16;
  const __bf16* Bp = Bpack + (size_t)(tn + lrow) * K + hlf * 16;

  v8f acc = {};
  for (int k = 0; k < K; k += 32) {
    v16bf a = *(const v16bf*)Ap;           // 32B -> 2x global_load_b128
    v16bf b = *(const v16bf*)Bp;
    __builtin_prefetch(Ap + 32, 0, 0);     // global_prefetch_b8 (speculative)
    __builtin_prefetch(Bp + 32, 0, 0);
    acc = __builtin_amdgcn_wmma_f32_16x16x32_bf16(
        false, a, false, b, (short)0, acc, false, false);
    Ap += 32;
    Bp += 32;
  }
  // C/D layout: VGPR v -> row v (lanes 0-15) / row v+8 (lanes 16-31)
  float* Cp = C + (size_t)(tm + hlf * 8) * Nc + tn + lrow;
#pragma unroll
  for (int v = 0; v < 8; ++v) Cp[(size_t)v * Nc] = acc[v];
}

// ---------------- attention scores: a_src/a_dst[n,h] = <h[n,h,:], att> ------
__global__ void att_scores(const float* __restrict__ h,
                           const float* __restrict__ att_s,
                           const float* __restrict__ att_d,
                           float* __restrict__ a_s, float* __restrict__ a_d,
                           int N, int H, int C) {
  int t = blockIdx.x * blockDim.x + threadIdx.x;
  if (t >= N * H) return;
  int hh = t % H;
  const float* hp = h + (size_t)t * C;
  const float* sv = att_s + hh * C;
  const float* dv = att_d + hh * C;
  float s0 = 0.f, d0 = 0.f;
#pragma unroll 4
  for (int c = 0; c < C; ++c) {
    float v = hp[c];
    s0 += v * sv[c];
    d0 += v * dv[c];
  }
  a_s[t] = s0;
  a_d[t] = d0;
}

// float atomic-max via int punning (lowers to native integer atomics)
static __device__ __forceinline__ void atomic_max_f32(float* addr, float v) {
  if (v >= 0.f) atomicMax((int*)addr, __float_as_int(v));
  else          atomicMin((unsigned int*)addr, __float_as_uint(v));
}

// ---------------- edge pass 1: leaky-relu logits + segment max --------------
__global__ void edge_logits_max(const long long* __restrict__ ei,
                                const float* __restrict__ a_s,
                                const float* __restrict__ a_d,
                                float* __restrict__ ebuf,
                                float* __restrict__ m, int E, int N, int H) {
  int t = blockIdx.x * blockDim.x + threadIdx.x;
  int tot = (E + N) * H;
  if (t >= tot) return;
  int e = t / H, hh = t - e * H;
  long long s, d;
  if (e < E) { s = ei[e]; d = ei[E + e]; } else { s = d = (long long)(e - E); }
  float v = a_s[s * H + hh] + a_d[d * H + hh];
  v = v > 0.f ? v : SLOPE * v;
  ebuf[t] = v;
  atomic_max_f32(&m[d * H + hh], v);
}

// ---------------- edge pass 2: exp(e - max) + segment sum -------------------
__global__ void edge_exp_sum(const long long* __restrict__ ei,
                             const float* __restrict__ m,
                             float* __restrict__ ebuf,
                             float* __restrict__ sbuf, int E, int N, int H) {
  int t = blockIdx.x * blockDim.x + threadIdx.x;
  int tot = (E + N) * H;
  if (t >= tot) return;
  int e = t / H, hh = t - e * H;
  long long d = (e < E) ? ei[E + e] : (long long)(e - E);
  float ex = __expf(ebuf[t] - m[d * H + hh]);
  ebuf[t] = ex;
  atomicAdd(&sbuf[d * H + hh], ex);
}

// ---------------- edge pass 3: alpha-weighted scatter-add -------------------
// one wave per (edge, head); 32 lanes x float2 = 64 channels
__global__ __launch_bounds__(256) void edge_scatter(
    const long long* __restrict__ ei, const float* __restrict__ hf,
    const float* __restrict__ ebuf, const float* __restrict__ sbuf,
    float* __restrict__ agg, int E, int N, int H) {
  int wave = blockIdx.x * (blockDim.x >> 5) + (threadIdx.x >> 5);
  int tot = (E + N) * H;
  if (wave >= tot) return;
  int lane = threadIdx.x & 31;
  int e = wave / H, hh = wave - e * H;
  long long s, d;
  if (e < E) { s = ei[e]; d = ei[E + e]; } else { s = d = (long long)(e - E); }
  float alpha = ebuf[wave] / (sbuf[d * H + hh] + 1e-16f);
  const float2 hv =
      *((const float2*)(hf + ((size_t)s * H + hh) * CCH) + lane);
  float* op = agg + ((size_t)d * H + hh) * CCH + lane * 2;
  atomicAdd(op,     hv.x * alpha);
  atomicAdd(op + 1, hv.y * alpha);
}

// ---------------- bias + ELU (elementwise, F = feature width) ---------------
__global__ void bias_elu(const float* __restrict__ in,
                         const float* __restrict__ bias,
                         float* __restrict__ out, int n, int F) {
  int i = blockIdx.x * blockDim.x + threadIdx.x;
  if (i >= n) return;
  float v = in[i] + bias[i % F];
  out[i] = v > 0.f ? v : __expf(v) - 1.f;
}

// ---------------------------------------------------------------------------
extern "C" void kernel_launch(void* const* d_in, const int* in_sizes, int n_in,
                              void* d_out, int out_size, void* d_ws,
                              size_t ws_size, hipStream_t stream) {
  const float*     x   = (const float*)d_in[0];
  const long long* ei  = (const long long*)d_in[1];  // int64 [2,E]
  const float*     W1  = (const float*)d_in[2];
  const float*     as1 = (const float*)d_in[3];
  const float*     ad1 = (const float*)d_in[4];
  const float*     b1  = (const float*)d_in[5];
  const float*     W2  = (const float*)d_in[6];
  const float*     as2 = (const float*)d_in[7];
  const float*     ad2 = (const float*)d_in[8];
  const float*     b2  = (const float*)d_in[9];
  float* out = (float*)d_out;
  (void)in_sizes; (void)n_in; (void)out_size; (void)ws_size;

  const int N = NN, E = EE, Etot = EE + NN;

  char* ws = (char*)d_ws;
  size_t off = 0;
  auto alloc = [&](size_t bytes) -> char* {
    char* p = ws + off;
    off = (off + bytes + 255) & ~(size_t)255;
    return p;
  };
  float*  h1    = (float*)alloc((size_t)N * 256 * 4);  // x@W1 ; later elu(agg1+b1)
  float*  agg1  = (float*)alloc((size_t)N * 256 * 4);  // layer-1 agg ; later h2
  float*  ebuf  = (float*)alloc((size_t)Etot * HEADS1 * 4);
  float*  a_s   = (float*)alloc((size_t)N * HEADS1 * 4);
  float*  a_d   = (float*)alloc((size_t)N * HEADS1 * 4);
  float*  mbuf  = (float*)alloc((size_t)N * HEADS1 * 4);
  float*  sbuf  = (float*)alloc((size_t)N * HEADS1 * 4);
  __bf16* apack = (__bf16*)alloc((size_t)N * 256 * 2);  // packed A (both GEMMs)
  __bf16* b1p   = (__bf16*)alloc((size_t)256 * 256 * 2);
  __bf16* b2p   = (__bf16*)alloc((size_t)256 * 64 * 2);

  // ================= operand packing (bf16, fragment order) =================
  pack_A_bf16<<<(N * 256 + 255) / 256, 256, 0, stream>>>(x, apack, N, 256);
  pack_B_bf16<<<(256 * 256 + 255) / 256, 256, 0, stream>>>(W1, b1p, 256, 256);
  pack_B_bf16<<<(256 * 64 + 255) / 256, 256, 0, stream>>>(W2, b2p, 256, 64);

  // ================= layer 1 (H=4, C=64, concat) =================
  {
    int waves = (N / 16) * (256 / 16);
    gemm_bf16_wmma<<<(waves + 7) / 8, 256, 0, stream>>>(apack, b1p, h1, N, 256,
                                                        256);
  }
  att_scores<<<(N * HEADS1 + 255) / 256, 256, 0, stream>>>(h1, as1, ad1, a_s,
                                                           a_d, N, HEADS1, CCH);
  fill_f32<<<2048, 256, 0, stream>>>(mbuf, -INFINITY, N * HEADS1);
  fill_f32<<<2048, 256, 0, stream>>>(sbuf, 0.f, N * HEADS1);
  fill_f32<<<8192, 256, 0, stream>>>(agg1, 0.f, N * 256);
  {
    int tot = Etot * HEADS1;
    edge_logits_max<<<(tot + 255) / 256, 256, 0, stream>>>(ei, a_s, a_d, ebuf,
                                                           mbuf, E, N, HEADS1);
    edge_exp_sum<<<(tot + 255) / 256, 256, 0, stream>>>(ei, mbuf, ebuf, sbuf,
                                                        E, N, HEADS1);
    edge_scatter<<<(tot + 7) / 8, 256, 0, stream>>>(ei, h1, ebuf, sbuf, agg1,
                                                    E, N, HEADS1);
  }
  bias_elu<<<(N * 256 + 255) / 256, 256, 0, stream>>>(agg1, b1, h1, N * 256,
                                                      256);

  // ================= layer 2 (H=1, C=64, mean==identity) =================
  float* h2 = agg1;  // reuse
  pack_A_bf16<<<(N * 256 + 255) / 256, 256, 0, stream>>>(h1, apack, N, 256);
  {
    int waves = (N / 16) * (64 / 16);
    gemm_bf16_wmma<<<(waves + 7) / 8, 256, 0, stream>>>(apack, b2p, h2, N, 256,
                                                        64);
  }
  att_scores<<<(N + 255) / 256, 256, 0, stream>>>(h2, as2, ad2, a_s, a_d, N, 1,
                                                  CCH);
  fill_f32<<<2048, 256, 0, stream>>>(mbuf, -INFINITY, N);
  fill_f32<<<2048, 256, 0, stream>>>(sbuf, 0.f, N);
  fill_f32<<<4096, 256, 0, stream>>>(out, 0.f, N * 64);
  {
    int tot = Etot;
    edge_logits_max<<<(tot + 255) / 256, 256, 0, stream>>>(ei, a_s, a_d, ebuf,
                                                           mbuf, E, N, 1);
    edge_exp_sum<<<(tot + 255) / 256, 256, 0, stream>>>(ei, mbuf, ebuf, sbuf,
                                                        E, N, 1);
    edge_scatter<<<(tot + 7) / 8, 256, 0, stream>>>(ei, h2, ebuf, sbuf, out, E,
                                                    N, 1);
  }
  bias_elu<<<(N * 64 + 255) / 256, 256, 0, stream>>>(out, b2, out, N * 64, 64);
}
